// AiLUT_29454885715985
// MI455X (gfx1250) — matrix operand order repaired and checked by reading, not compiled, not run on hardware
//
#include <hip/hip_runtime.h>
#include <hip/hip_bf16.h>

typedef __attribute__((ext_vector_type(16))) _Float16 v16h;
typedef __attribute__((ext_vector_type(8)))  float    v8f;

#define BATCH 8
#define VCUBE 35937            // 33*33*33
#define LUTSZ (3*VCUBE)        // 107811

// ---------------------------------------------------------------------------
// 1) Antialiased bilinear resize 1024 -> 256 (scale 4, half-pixel centers).
// src center = 4*oy + 1.5 ; triangle kernel radius 4 -> 8 taps, weights
// [0.125,0.375,0.625,0.875,0.875,0.625,0.375,0.125] per dim, per-dim
// renormalized at borders (matches jax.image.resize antialias=True).
// ---------------------------------------------------------------------------
__global__ __launch_bounds__(256)
void k_resize(const float* __restrict__ x, float* __restrict__ y) {
  const int NOUT = BATCH * 3 * 256 * 256;
  int idx = blockIdx.x * blockDim.x + threadIdx.x;
  if (idx >= NOUT) return;
  int ox = idx & 255;
  int oy = (idx >> 8) & 255;
  int bc = idx >> 16;                       // b*3 + c
  const float* src = x + (size_t)bc * 1024 * 1024;
  const float w8[8] = {0.125f,0.375f,0.625f,0.875f,0.875f,0.625f,0.375f,0.125f};
  int iy0 = 4 * oy - 2;
  int ix0 = 4 * ox - 2;
  float wy[8], wx[8], sy = 0.f, sx = 0.f;
  #pragma unroll
  for (int j = 0; j < 8; ++j) {
    int iy = iy0 + j;
    wy[j] = (iy >= 0 && iy < 1024) ? w8[j] : 0.f; sy += wy[j];
    int ix = ix0 + j;
    wx[j] = (ix >= 0 && ix < 1024) ? w8[j] : 0.f; sx += wx[j];
  }
  float acc = 0.f;
  #pragma unroll
  for (int j = 0; j < 8; ++j) {
    if (wy[j] == 0.f) continue;
    int iy = iy0 + j;
    float rowacc = 0.f;
    #pragma unroll
    for (int i = 0; i < 8; ++i) {
      if (wx[i] == 0.f) continue;
      rowacc += wx[i] * src[(size_t)iy * 1024 + (ix0 + i)];
    }
    acc += wy[j] * rowacc;
  }
  y[idx] = acc / (sy * sx);
}

// ---------------------------------------------------------------------------
// 2) Implicit-GEMM 3x3/stride-2/pad-1 conv via v_wmma_f32_16x16x32_f16.
// Block = 4 waves. Each block owns one 16(co) x 16(pixel) output tile; the
// 4 waves split the K-chunks (K = CI*9, 32 per chunk) round-robin, each with
// a private LDS A/B slab (no block barrier in the K loop; intra-wave DScnt
// ordering is enough). Staging is branchless: clamped addresses -> all 16
// global_load_b32 issued back-to-back, masks applied on the f32->f16 LDS
// store. Partial C fragments reduced through LDS; wave 0 does bias + leaky.
// ---------------------------------------------------------------------------
template<int CI, int CO, int HIN, int WIN>
__global__ __launch_bounds__(128)
void k_conv_wmma(const float* __restrict__ in, const float* __restrict__ wgt,
                 const float* __restrict__ bias, float* __restrict__ out) {
  constexpr int HOUT = HIN / 2, WOUT = WIN / 2;
  constexpr int K = CI * 9;
  constexpr int N = BATCH * HOUT * WOUT;
  constexpr int KT = (K + 31) / 32;
  static_assert(N % 16 == 0, "tile never crosses N bound");
  __shared__ _Float16 sA[4][16 * 32];   // per-wave [m][k]
  __shared__ _Float16 sB[4][32 * 16];   // per-wave [k][n]
  __shared__ float    sC[4][8 * 32];    // per-wave partial C
  const int t = threadIdx.x & 31;
  const int w = threadIdx.x >> 5;
  const int nTile  = blockIdx.x;
  const int coTile = blockIdx.y;
  const int n0 = nTile * 16;
  _Float16* sAw = &sA[w][0];
  _Float16* sBw = &sB[w][0];

  // per-lane invariants
  const int lane_n = t & 15;
  const int half   = t >> 4;
  const int arow   = t >> 1;               // A: lane covers half a row
  const int acol0  = (t & 1) * 16;
  const float* wrow = wgt + (size_t)(coTile * 16 + arow) * K;

  v8f acc = {};
  for (int kt = w; kt < KT; kt += 4) {
    const int k0 = kt * 32;
    // ---- stage A: lane t -> row t/2, cols acol0..acol0+15 ----
    {
      float av[16];
      #pragma unroll
      for (int j = 0; j < 16; ++j) {
        int k  = k0 + acol0 + j;
        int kc = (k < K) ? k : (K - 1);       // clamped, always in-bounds
        av[j] = wrow[kc];
      }
      #pragma unroll
      for (int j = 0; j < 16; ++j) {
        int k = k0 + acol0 + j;
        sAw[arow * 32 + acol0 + j] = (_Float16)((k < K) ? av[j] : 0.f);
      }
    }
    // ---- stage B: lane t -> im2col row kk = t (fixed ci,dy,dx per lane) ----
    {
      int k  = k0 + t;
      bool kok = (k < K);
      int kc = kok ? k : 0;
      int ci = kc / 9, rm = kc % 9;
      int dy = rm / 3, dx = rm % 3;
      float bv[16];
      bool  bok[16];
      #pragma unroll
      for (int j = 0; j < 16; ++j) {
        int n  = n0 + j;
        int b  = n / (HOUT * WOUT), p = n % (HOUT * WOUT);
        int oy = p / WOUT, ox = p % WOUT;
        int iy = oy * 2 + dy - 1, ix = ox * 2 + dx - 1;
        bool ok = kok && iy >= 0 && iy < HIN && ix >= 0 && ix < WIN;
        int iyc = iy < 0 ? 0 : (iy >= HIN ? HIN - 1 : iy);
        int ixc = ix < 0 ? 0 : (ix >= WIN ? WIN - 1 : ix);
        bv[j]  = in[(((size_t)b * CI + ci) * HIN + iyc) * WIN + ixc];
        bok[j] = ok;
      }
      #pragma unroll
      for (int j = 0; j < 16; ++j)
        sBw[t * 16 + j] = (_Float16)(bok[j] ? bv[j] : 0.f);
    }
    // ---- pack fragments (ISA 7.12.2, wave32) and accumulate ----
    v16h a = {}, bf = {};
    #pragma unroll
    for (int i = 0; i < 4; ++i) {        // A VGPR 0..3: K = half*8 + 2i,2i+1
      int kA = half * 8 + 2 * i;
      a[2 * i]     = sAw[lane_n * 32 + kA];
      a[2 * i + 1] = sAw[lane_n * 32 + kA + 1];
    }
    #pragma unroll
    for (int i = 4; i < 8; ++i) {        // A VGPR 4..7: K = 16 + half*8 + ...
      int kA = 16 + half * 8 + 2 * (i - 4);
      a[2 * i]     = sAw[lane_n * 32 + kA];
      a[2 * i + 1] = sAw[lane_n * 32 + kA + 1];
    }
    #pragma unroll
    for (int i = 0; i < 8; ++i) {        // B VGPR i: K = half*8 + 2i (+8 for i>=4), +1
      int kB = half * 8 + 2 * i + (i >= 4 ? 8 : 0);
      bf[2 * i]     = sBw[kB * 16 + lane_n];
      bf[2 * i + 1] = sBw[(kB + 1) * 16 + lane_n];
    }
    acc = __builtin_amdgcn_wmma_f32_16x16x32_f16(
        false, a, false, bf, (short)0, acc, false, false);
  }
  // ---- cross-wave K reduction + epilogue ----
  #pragma unroll
  for (int r = 0; r < 8; ++r) sC[w][r * 32 + t] = acc[r];
  __syncthreads();
  if (w == 0) {
    int n  = n0 + lane_n;
    int b  = n / (HOUT * WOUT), p = n % (HOUT * WOUT);
    int oy = p / WOUT, ox = p % WOUT;
    #pragma unroll
    for (int r = 0; r < 8; ++r) {
      float v = sC[0][r * 32 + t] + sC[1][r * 32 + t] +
                sC[2][r * 32 + t] + sC[3][r * 32 + t];
      int co = coTile * 16 + half * 8 + r;
      v += bias[co];
      v = (v >= 0.f) ? v : 0.2f * v;
      out[(((size_t)b * CO + co) * HOUT + oy) * WOUT + ox] = v;
    }
  }
}

// ---------------------------------------------------------------------------
// 3) Instance norm (per b,c over HxW), in place. Population variance.
// ---------------------------------------------------------------------------
template<int HW>
__global__ __launch_bounds__(256)
void k_inorm(float* __restrict__ x, const float* __restrict__ g,
             const float* __restrict__ be, int C) {
  int bc = blockIdx.x;              // b*C + c
  int c  = bc % C;
  float* p = x + (size_t)bc * HW;
  __shared__ float s1[256], s2[256];
  float sum = 0.f, sq = 0.f;
  for (int i = threadIdx.x; i < HW; i += 256) {
    float v = p[i]; sum += v; sq += v * v;
  }
  s1[threadIdx.x] = sum; s2[threadIdx.x] = sq;
  __syncthreads();
  for (int s = 128; s > 0; s >>= 1) {
    if (threadIdx.x < s) { s1[threadIdx.x] += s1[threadIdx.x + s];
                           s2[threadIdx.x] += s2[threadIdx.x + s]; }
    __syncthreads();
  }
  float mean = s1[0] * (1.f / HW);
  float var  = s2[0] * (1.f / HW) - mean * mean;
  float scale = g[c] * rsqrtf(var + 1e-5f);
  float shift = be[c] - mean * scale;
  for (int i = threadIdx.x; i < HW; i += 256) p[i] = p[i] * scale + shift;
}

// ---------------------------------------------------------------------------
// 4) FC: weights[b,o] = dot(codes[b,:8192], wg_w[o,:]) + wg_b[o]
// ---------------------------------------------------------------------------
__global__ __launch_bounds__(256)
void k_fc(const float* __restrict__ codes, const float* __restrict__ wgw,
          const float* __restrict__ wgb, float* __restrict__ w_ws,
          float* __restrict__ w_out) {
  int bo = blockIdx.x;              // b*3 + o
  int b = bo / 3, o = bo % 3;
  __shared__ float s[256];
  float sum = 0.f;
  for (int i = threadIdx.x; i < 8192; i += 256)
    sum += codes[(size_t)b * 8192 + i] * wgw[(size_t)o * 8192 + i];
  s[threadIdx.x] = sum;
  __syncthreads();
  for (int st = 128; st > 0; st >>= 1) {
    if (threadIdx.x < st) s[threadIdx.x] += s[threadIdx.x + st];
    __syncthreads();
  }
  if (threadIdx.x == 0) {
    float v = s[0] + wgb[o];
    w_ws[bo] = v;
    w_out[bo] = v;
  }
}

// ---------------------------------------------------------------------------
// 5) LUT synthesis: luts[b, j] = sum_i weights[b,i] * lut_w[j, i]
// ---------------------------------------------------------------------------
__global__ __launch_bounds__(256)
void k_lut(const float* __restrict__ w, const float* __restrict__ lut_w,
           float* __restrict__ luts) {
  int idx = blockIdx.x * blockDim.x + threadIdx.x;
  if (idx >= BATCH * LUTSZ) return;
  int b = idx / LUTSZ, j = idx % LUTSZ;
  const float* wb = w + b * 3;
  luts[idx] = wb[0] * lut_w[(size_t)j * 3 + 0] +
              wb[1] * lut_w[(size_t)j * 3 + 1] +
              wb[2] * lut_w[(size_t)j * 3 + 2];
}

__global__ void k_copy(const float* __restrict__ src, float* __restrict__ dst, int n) {
  int i = blockIdx.x * blockDim.x + threadIdx.x;
  if (i < n) dst[i] = src[i];
}

// ---------------------------------------------------------------------------
// 6) Trilinear LUT apply (memory-bound streaming pass; LUTs sit in L2).
// Uniform vertices k/32 -> idx = clamp(floor(32q), 0, 31), d = 32q - idx.
// ---------------------------------------------------------------------------
__global__ __launch_bounds__(256)
void k_ailut(const float* __restrict__ x, const float* __restrict__ luts,
             float* __restrict__ out) {
  const int HW = 1024 * 1024;
  int idx = blockIdx.x * blockDim.x + threadIdx.x;
  if (idx >= BATCH * HW) return;
  int b = idx >> 20, p = idx & (HW - 1);
  const float* xb = x + (size_t)b * 3 * HW;
  int   id[3];
  float d[3];
  #pragma unroll
  for (int c = 0; c < 3; ++c) {
    float v = xb[(size_t)c * HW + p];
    int i = (int)floorf(v * 32.f);
    i = i < 0 ? 0 : (i > 31 ? 31 : i);
    id[c] = i;
    d[c]  = v * 32.f - (float)i;
  }
  int   ir = id[0], ig = id[1], ib = id[2];
  float dr = d[0],  dg = d[1],  db = d[2];
  const float* lb = luts + (size_t)b * LUTSZ;
  float o0 = 0.f, o1 = 0.f, o2 = 0.f;
  #pragma unroll
  for (int cb = 0; cb < 2; ++cb)
    #pragma unroll
    for (int cg = 0; cg < 2; ++cg)
      #pragma unroll
      for (int cr = 0; cr < 2; ++cr) {
        float wv = (cb ? db : 1.f - db) * (cg ? dg : 1.f - dg) * (cr ? dr : 1.f - dr);
        int lin = ((ib + cb) * 33 + (ig + cg)) * 33 + (ir + cr);
        o0 += wv * lb[lin];
        o1 += wv * lb[VCUBE + lin];
        o2 += wv * lb[2 * VCUBE + lin];
      }
  float* ob = out + (size_t)b * 3 * HW;
  ob[p] = o0; ob[HW + p] = o1; ob[2 * HW + p] = o2;
}

// ---------------------------------------------------------------------------
extern "C" void kernel_launch(void* const* d_in, const int* in_sizes, int n_in,
                              void* d_out, int out_size, void* d_ws, size_t ws_size,
                              hipStream_t stream) {
  // setup_inputs() dict order
  const float* x    = (const float*)d_in[0];
  const float* cw0  = (const float*)d_in[1];  const float* cb0 = (const float*)d_in[2];
  const float* cw1  = (const float*)d_in[3];  const float* cb1 = (const float*)d_in[4];
  const float* cw2  = (const float*)d_in[5];  const float* cb2 = (const float*)d_in[6];
  const float* cw3  = (const float*)d_in[7];  const float* cb3 = (const float*)d_in[8];
  const float* cw4  = (const float*)d_in[9];  const float* cb4 = (const float*)d_in[10];
  const float* g0   = (const float*)d_in[11]; const float* be0 = (const float*)d_in[12];
  const float* g1   = (const float*)d_in[13]; const float* be1 = (const float*)d_in[14];
  const float* g2   = (const float*)d_in[15]; const float* be2 = (const float*)d_in[16];
  const float* g3   = (const float*)d_in[17]; const float* be3 = (const float*)d_in[18];
  const float* wgw  = (const float*)d_in[19]; const float* wgb = (const float*)d_in[20];
  const float* lutw = (const float*)d_in[21]; const float* vert = (const float*)d_in[22];

  // workspace layout (floats): ~22.3 MB total
  float* ws   = (float*)d_ws;
  float* rbuf = ws;                         // 8*3*256*256    = 1572864
  float* a0   = rbuf + 1572864;             // 8*16*128*128   = 2097152
  float* a1   = a0 + 2097152;               // 8*32*64*64     = 1048576
  float* a2   = a1 + 1048576;               // 8*64*32*32     =  524288
  float* a3   = a2 + 524288;                // 8*128*16*16    =  262144
  float* a4   = a3 + 262144;                // 8*128*8*8      =   65536
  float* wbuf = a4 + 65536;                 // 24

  // d_out: [outs | weights | luts | vertices]
  float* out_img = (float*)d_out;
  float* out_w   = out_img + (size_t)BATCH * 3 * 1024 * 1024;  // +25165824
  float* out_lut = out_w + 24;
  float* out_v   = out_lut + (size_t)BATCH * LUTSZ;            // +862488

  { int n = BATCH * 3 * 256 * 256;
    k_resize<<<(n + 255) / 256, 256, 0, stream>>>(x, rbuf); }

  { int nt = (BATCH * 128 * 128) / 16;      // conv0: 3->16, 256->128
    k_conv_wmma<3, 16, 256, 256><<<dim3(nt, 1), 128, 0, stream>>>(rbuf, cw0, cb0, a0);
    k_inorm<128 * 128><<<BATCH * 16, 256, 0, stream>>>(a0, g0, be0, 16); }

  { int nt = (BATCH * 64 * 64) / 16;        // conv1: 16->32, 128->64
    k_conv_wmma<16, 32, 128, 128><<<dim3(nt, 2), 128, 0, stream>>>(a0, cw1, cb1, a1);
    k_inorm<64 * 64><<<BATCH * 32, 256, 0, stream>>>(a1, g1, be1, 32); }

  { int nt = (BATCH * 32 * 32) / 16;        // conv2: 32->64, 64->32
    k_conv_wmma<32, 64, 64, 64><<<dim3(nt, 4), 128, 0, stream>>>(a1, cw2, cb2, a2);
    k_inorm<32 * 32><<<BATCH * 64, 256, 0, stream>>>(a2, g2, be2, 64); }

  { int nt = (BATCH * 16 * 16) / 16;        // conv3: 64->128, 32->16
    k_conv_wmma<64, 128, 32, 32><<<dim3(nt, 8), 128, 0, stream>>>(a2, cw3, cb3, a3);
    k_inorm<16 * 16><<<BATCH * 128, 256, 0, stream>>>(a3, g3, be3, 128); }

  { int nt = (BATCH * 8 * 8) / 16;          // conv4: 128->128, 16->8 (no inorm)
    k_conv_wmma<128, 128, 16, 16><<<dim3(nt, 8), 128, 0, stream>>>(a3, cw4, cb4, a4); }

  k_fc<<<24, 256, 0, stream>>>(a4, wgw, wgb, wbuf, out_w);

  { int n = BATCH * LUTSZ;
    k_lut<<<(n + 255) / 256, 256, 0, stream>>>(wbuf, lutw, out_lut); }

  k_copy<<<1, 128, 0, stream>>>(vert, out_v, 99);

  { int n = BATCH * 1024 * 1024;
    k_ailut<<<(n + 255) / 256, 256, 0, stream>>>(x, out_lut, out_img); }
}
